// ScanCausalSelfAttention_77283641524842
// MI455X (gfx1250) — matrix operand order repaired and checked by reading, not compiled, not run on hardware
//
#include <hip/hip_runtime.h>
#include <cstddef>

// Problem constants (B=2, T=2048, D=1024, H=16, head_dim=64)
#define NB 2
#define TT 2048
#define DD 1024
#define NH 16
#define DH 64
#define MROWS (NB * TT)   // 4096

typedef __attribute__((ext_vector_type(16))) __bf16 v16bf;
typedef __attribute__((ext_vector_type(8)))  float  v8f;
typedef __attribute__((ext_vector_type(4)))  unsigned int u32x4;
typedef unsigned short u16;

union Frag { u32x4 q[2]; v16bf v; };

__device__ __forceinline__ u16 f2bf(float f) {
  union { float f; unsigned int u; } x; x.f = f;
  unsigned int r = x.u + 0x7FFFu + ((x.u >> 16) & 1u);  // RNE
  return (u16)(r >> 16);
}

// A-fragment: 16x32 bf16 (MxK), row-major source, leading dim lda (elements).
// Lanes 0-15 -> M=lane, K in [g*8,g*8+8) U [16+g*8, 16+g*8+8)   (ISA 7.12.2)
__device__ __forceinline__ v16bf load_a(const u16* __restrict__ p, int lda, int lane) {
  int m = lane & 15, g = lane >> 4;
  const u16* q = p + (long)m * lda + g * 8;
  Frag f;
  f.q[0] = *(const u32x4*)(q);
  f.q[1] = *(const u32x4*)(q + 16);
  return f.v;
}

// B-fragment: 32x16 bf16 (KxN) from transposed ("Bt" = [N,K] row-major) source.
// Lane holds column n = lane&15; K in [g*8,g*8+8) U [16+g*8, 16+g*8+8) (matches A).
__device__ __forceinline__ v16bf load_bt(const u16* __restrict__ p, long ldb, int lane) {
  int n = lane & 15, g = lane >> 4;
  const u16* q = p + (long)n * ldb + g * 8;
  Frag f;
  f.q[0] = *(const u32x4*)(q);
  f.q[1] = *(const u32x4*)(q + 16);
  return f.v;
}

#define WMMA_BF16(a, b, c) \
  __builtin_amdgcn_wmma_f32_16x16x32_bf16(false, (a), false, (b), (short)0, (c), false, false)

// ---------------- prep kernels ----------------
__global__ void k_cvt(const float* __restrict__ in, u16* __restrict__ out, int n) {
  int i = blockIdx.x * blockDim.x + threadIdx.x;
  int stride = gridDim.x * blockDim.x;
  for (; i < n; i += stride) out[i] = f2bf(in[i]);
}

// W: [K,N] f32 row-major  ->  Wt: [N,K] bf16 row-major
__global__ void k_transpose_cvt(const float* __restrict__ W, u16* __restrict__ Wt,
                                int K, int N) {
  long total = (long)K * N;
  long i = (long)blockIdx.x * blockDim.x + threadIdx.x;
  long stride = (long)gridDim.x * blockDim.x;
  for (; i < total; i += stride) {
    long n = i / K, k = i - n * K;
    Wt[i] = f2bf(W[k * (long)N + n]);
  }
}

// ---------------- QKV GEMM: [4096,1024] x [1024,3072] ----------------
// Writes Q (scaled 0.125) and K as [B,H,T,64] bf16, V transposed [B,H,64,T] bf16.
// Software-pipelined: fragments for K-step kk+32 are loaded before the WMMAs of kk.
__global__ void __launch_bounds__(128)
k_gemm_qkv(const u16* __restrict__ A, const u16* __restrict__ Bt,
           u16* __restrict__ Qb, u16* __restrict__ Kb, u16* __restrict__ Vt) {
  int lane = threadIdx.x & 31, wave = threadIdx.x >> 5;
  int m0 = blockIdx.y * 64 + wave * 16;
  int n0 = blockIdx.x * 64;
  v8f c[4] = {};
  const u16* Ap = A + (long)m0 * DD;
  const u16* Bp0 = Bt + (long)(n0 + 0)  * DD;
  const u16* Bp1 = Bt + (long)(n0 + 16) * DD;
  const u16* Bp2 = Bt + (long)(n0 + 32) * DD;
  const u16* Bp3 = Bt + (long)(n0 + 48) * DD;

  v16bf ac = load_a(Ap, DD, lane);
  v16bf b0 = load_bt(Bp0, DD, lane);
  v16bf b1 = load_bt(Bp1, DD, lane);
  v16bf b2 = load_bt(Bp2, DD, lane);
  v16bf b3 = load_bt(Bp3, DD, lane);

  for (int kk = 0; kk < DD - 32; kk += 32) {
    int kn = kk + 32;
    __builtin_prefetch(Ap + kn + 32, 0, 1);
    v16bf an  = load_a(Ap + kn, DD, lane);
    v16bf n0f = load_bt(Bp0 + kn, DD, lane);
    v16bf n1f = load_bt(Bp1 + kn, DD, lane);
    v16bf n2f = load_bt(Bp2 + kn, DD, lane);
    v16bf n3f = load_bt(Bp3 + kn, DD, lane);
    c[0] = WMMA_BF16(ac, b0, c[0]);
    c[1] = WMMA_BF16(ac, b1, c[1]);
    c[2] = WMMA_BF16(ac, b2, c[2]);
    c[3] = WMMA_BF16(ac, b3, c[3]);
    ac = an; b0 = n0f; b1 = n1f; b2 = n2f; b3 = n3f;
  }
  c[0] = WMMA_BF16(ac, b0, c[0]);
  c[1] = WMMA_BF16(ac, b1, c[1]);
  c[2] = WMMA_BF16(ac, b2, c[2]);
  c[3] = WMMA_BF16(ac, b3, c[3]);

  int g = lane >> 4, nl = lane & 15;
#pragma unroll
  for (int nt = 0; nt < 4; ++nt) {
#pragma unroll
    for (int r = 0; r < 8; ++r) {
      int n = n0 + nt * 16 + nl;          // [0, 3072)
      int row = m0 + r + 8 * g;           // [0, 4096) = b*T + t
      float v = c[nt][r];
      int which = n >> 10;                // 0=Q 1=K 2=V
      int ccol = n & 1023;
      int h = ccol >> 6, dd = ccol & 63;
      int bb = row >> 11, t = row & 2047;
      long bh = (long)bb * NH + h;
      if (which == 0)      Qb[(bh * TT + t) * DH + dd] = f2bf(v * 0.125f);
      else if (which == 1) Kb[(bh * TT + t) * DH + dd] = f2bf(v);
      else                 Vt[(bh * DH + dd) * TT + t] = f2bf(v);
    }
  }
}

// ---------------- flash attention: one wave = 16 q-rows of one (b,h) ----------------
__global__ void __launch_bounds__(128)
k_attn(const u16* __restrict__ Qb, const u16* __restrict__ Kb,
       const u16* __restrict__ Vt, u16* __restrict__ AOut) {
  __shared__ __align__(16) u16 pbuf[4][16 * 32];
  int lane = threadIdx.x & 31, wave = threadIdx.x >> 5;
  int gid = blockIdx.x * 4 + wave;
  int qt = gid & 127;           // q tile [0,128)
  int h = (gid >> 7) & 15;
  int b = gid >> 11;
  long bh = (long)b * NH + h;
  const u16* Qp = Qb + (bh * TT + (long)qt * 16) * DH;
  const u16* Kp = Kb + bh * TT * DH;
  const u16* Vp = Vt + bh * DH * TT;

  v16bf a0 = load_a(Qp, DH, lane);       // head-dim 0..31
  v16bf a1 = load_a(Qp + 32, DH, lane);  // head-dim 32..63
  v8f o[4] = {};
  float mrow[8], lrow[8];
#pragma unroll
  for (int r = 0; r < 8; ++r) { mrow[r] = -1e30f; lrow[r] = 0.f; }
  int g = lane >> 4, nl = lane & 15;
  u16* pl = pbuf[wave];
  int nch = (qt >> 1) + 1;               // causal: keys up to qt*16+15

  // pre-load K fragments for chunk 0
  v16bf kb0 = load_bt(Kp, DH, lane);
  v16bf kb1 = load_bt(Kp + 32, DH, lane);
  v16bf kb2 = load_bt(Kp + (long)16 * DH, DH, lane);
  v16bf kb3 = load_bt(Kp + (long)16 * DH + 32, DH, lane);

  for (int ch = 0; ch < nch; ++ch) {
    int k0 = ch * 32;
    v8f s0 = {}, s1 = {};
    s0 = WMMA_BF16(a0, kb0, s0);
    s0 = WMMA_BF16(a1, kb1, s0);
    s1 = WMMA_BF16(a0, kb2, s1);
    s1 = WMMA_BF16(a1, kb3, s1);

    // issue this chunk's V fragments + next chunk's K fragments now, so the
    // softmax VALU block below hides their latency
    v16bf bv0 = load_bt(Vp + (long)0  * TT + k0, (long)TT, lane);
    v16bf bv1 = load_bt(Vp + (long)16 * TT + k0, (long)TT, lane);
    v16bf bv2 = load_bt(Vp + (long)32 * TT + k0, (long)TT, lane);
    v16bf bv3 = load_bt(Vp + (long)48 * TT + k0, (long)TT, lane);
    long kn = (ch + 1 < nch) ? (long)(k0 + 32) : (long)k0;  // clamp: safe re-load
    kb0 = load_bt(Kp + kn * DH, DH, lane);
    kb1 = load_bt(Kp + kn * DH + 32, DH, lane);
    kb2 = load_bt(Kp + (kn + 16) * DH, DH, lane);
    kb3 = load_bt(Kp + (kn + 16) * DH + 32, DH, lane);

    float alpha[8];
#pragma unroll
    for (int r = 0; r < 8; ++r) {
      int row = qt * 16 + r + 8 * g;
      float v0 = ((k0 + nl) <= row)      ? s0[r] : -1e30f;
      float v1 = ((k0 + 16 + nl) <= row) ? s1[r] : -1e30f;
      float mx = fmaxf(v0, v1);
      mx = fmaxf(mx, __shfl_xor(mx, 1, 32));
      mx = fmaxf(mx, __shfl_xor(mx, 2, 32));
      mx = fmaxf(mx, __shfl_xor(mx, 4, 32));
      mx = fmaxf(mx, __shfl_xor(mx, 8, 32));   // row-max over 16-lane half group
      float mnew = fmaxf(mrow[r], mx);
      float al = __expf(mrow[r] - mnew);
      float p0 = __expf(v0 - mnew);
      float p1 = __expf(v1 - mnew);
      float sr = p0 + p1;
      sr += __shfl_xor(sr, 1, 32);
      sr += __shfl_xor(sr, 2, 32);
      sr += __shfl_xor(sr, 4, 32);
      sr += __shfl_xor(sr, 8, 32);
      lrow[r] = lrow[r] * al + sr;
      mrow[r] = mnew;
      alpha[r] = al;
      u16* pr = pl + (r + 8 * g) * 32;         // P tile 16x32 in LDS (C -> A relayout)
      pr[nl] = f2bf(p0);
      pr[16 + nl] = f2bf(p1);
    }
#pragma unroll
    for (int nt = 0; nt < 4; ++nt)
#pragma unroll
      for (int r = 0; r < 8; ++r) o[nt][r] *= alpha[r];

    asm volatile("s_wait_dscnt 0" ::: "memory");
    v16bf pa = load_a(pl, 32, lane);           // P as 16x32 A-fragment
    o[0] = WMMA_BF16(pa, bv0, o[0]);
    o[1] = WMMA_BF16(pa, bv1, o[1]);
    o[2] = WMMA_BF16(pa, bv2, o[2]);
    o[3] = WMMA_BF16(pa, bv3, o[3]);
  }

#pragma unroll
  for (int r = 0; r < 8; ++r) {
    float inv = 1.0f / lrow[r];
    int row = qt * 16 + r + 8 * g;
    long orow = ((long)b * TT + row) * DD + h * DH;
#pragma unroll
    for (int nt = 0; nt < 4; ++nt)
      AOut[orow + nt * 16 + nl] = f2bf(o[nt][r] * inv);
  }
}

// ---------------- output projection: [4096,1024] x [1024,1024] -> f32 ----------------
__global__ void __launch_bounds__(128)
k_gemm_proj(const u16* __restrict__ A, const u16* __restrict__ Bt,
            float* __restrict__ out) {
  int lane = threadIdx.x & 31, wave = threadIdx.x >> 5;
  int m0 = blockIdx.y * 64 + wave * 16;
  int n0 = blockIdx.x * 64;
  v8f c[4] = {};
  const u16* Ap = A + (long)m0 * DD;
  const u16* Bp0 = Bt + (long)(n0 + 0)  * DD;
  const u16* Bp1 = Bt + (long)(n0 + 16) * DD;
  const u16* Bp2 = Bt + (long)(n0 + 32) * DD;
  const u16* Bp3 = Bt + (long)(n0 + 48) * DD;

  v16bf ac = load_a(Ap, DD, lane);
  v16bf b0 = load_bt(Bp0, DD, lane);
  v16bf b1 = load_bt(Bp1, DD, lane);
  v16bf b2 = load_bt(Bp2, DD, lane);
  v16bf b3 = load_bt(Bp3, DD, lane);

  for (int kk = 0; kk < DD - 32; kk += 32) {
    int kn = kk + 32;
    __builtin_prefetch(Ap + kn + 32, 0, 1);
    v16bf an  = load_a(Ap + kn, DD, lane);
    v16bf n0f = load_bt(Bp0 + kn, DD, lane);
    v16bf n1f = load_bt(Bp1 + kn, DD, lane);
    v16bf n2f = load_bt(Bp2 + kn, DD, lane);
    v16bf n3f = load_bt(Bp3 + kn, DD, lane);
    c[0] = WMMA_BF16(ac, b0, c[0]);
    c[1] = WMMA_BF16(ac, b1, c[1]);
    c[2] = WMMA_BF16(ac, b2, c[2]);
    c[3] = WMMA_BF16(ac, b3, c[3]);
    ac = an; b0 = n0f; b1 = n1f; b2 = n2f; b3 = n3f;
  }
  c[0] = WMMA_BF16(ac, b0, c[0]);
  c[1] = WMMA_BF16(ac, b1, c[1]);
  c[2] = WMMA_BF16(ac, b2, c[2]);
  c[3] = WMMA_BF16(ac, b3, c[3]);

  int g = lane >> 4, nl = lane & 15;
#pragma unroll
  for (int nt = 0; nt < 4; ++nt)
#pragma unroll
    for (int r = 0; r < 8; ++r)
      out[(long)(m0 + r + 8 * g) * DD + n0 + nt * 16 + nl] = c[nt][r];
}

extern "C" void kernel_launch(void* const* d_in, const int* in_sizes, int n_in,
                              void* d_out, int out_size, void* d_ws, size_t ws_size,
                              hipStream_t stream) {
  const float* X     = (const float*)d_in[0];
  const float* Wqkv  = (const float*)d_in[1];
  const float* Wproj = (const float*)d_in[2];
  float* out = (float*)d_out;

  char* ws = (char*)d_ws;
  u16* Xbf    = (u16*)ws; ws += (size_t)MROWS * DD * 2;        // 8 MB
  u16* WqkvT  = (u16*)ws; ws += (size_t)3 * DD * DD * 2;       // 6 MB
  u16* WprojT = (u16*)ws; ws += (size_t)DD * DD * 2;           // 2 MB
  u16* Qb     = (u16*)ws; ws += (size_t)NB * NH * TT * DH * 2; // 8 MB
  u16* Kb     = (u16*)ws; ws += (size_t)NB * NH * TT * DH * 2; // 8 MB
  u16* Vt     = (u16*)ws; ws += (size_t)NB * NH * TT * DH * 2; // 8 MB
  u16* AOut   = (u16*)ws;                                      // 8 MB

  k_cvt<<<1024, 256, 0, stream>>>(X, Xbf, MROWS * DD);
  k_transpose_cvt<<<1024, 256, 0, stream>>>(Wqkv, WqkvT, DD, 3 * DD);
  k_transpose_cvt<<<1024, 256, 0, stream>>>(Wproj, WprojT, DD, DD);

  dim3 g1(3 * DD / 64, MROWS / 64);   // 48 x 64
  k_gemm_qkv<<<g1, 128, 0, stream>>>(Xbf, WqkvT, Qb, Kb, Vt);

  k_attn<<<(NB * NH * (TT / 16)) / 4, 128, 0, stream>>>(Qb, Kb, Vt, AOut);

  dim3 g2(DD / 64, MROWS / 64);       // 16 x 64
  k_gemm_proj<<<g2, 128, 0, stream>>>(AOut, WprojT, out);
}
